// SparseConvNet_8211977470296
// MI455X (gfx1250) — matrix operand order repaired and checked by reading, not compiled, not run on hardware
//
#include <hip/hip_runtime.h>
#include <hip/hip_bf16.h>

typedef __attribute__((ext_vector_type(16))) __bf16 v16bf;
typedef __attribute__((ext_vector_type(8)))  __bf16 v8bf;
typedef __attribute__((ext_vector_type(8)))  float  v8f;

#define GS_NPTS 65536

// ---------------------------------------------------------------------------
// Pack one layer's weights (Cout,Cin,3,3,3 fp32) into WMMA B-matrix layout:
//   wpk[((t*2 + ntile)*32 + lane)*16 + e]  (bf16)
//   lane<16 : column N = lane,      elements e -> K = e<8 ? e : e+8      (in-ch 0..7,16..23)
//   lane>=16: column N = lane-16,   elements e -> K = e<8 ? e+8 : e+16   (in-ch 8..15,24..31)
// Pad channels (co>=Cout or ci>=Cin) are zero so outputs stay exactly 0.
// Also emits padded scale/bias: spb[0..31]=scale, spb[32..63]=bias.
// ---------------------------------------------------------------------------
__global__ void pack_weights_k(const float* __restrict__ w,
                               const float* __restrict__ sc,
                               const float* __restrict__ bi,
                               __bf16* __restrict__ wpk,
                               float* __restrict__ spb,
                               int Cin, int Cout)
{
    int i = blockIdx.x * blockDim.x + threadIdx.x;
    if (i < 27 * 2 * 32 * 16) {
        int e    = i & 15;
        int lane = (i >> 4) & 31;
        int nt   = (i >> 9) & 1;
        int t    = i >> 10;                 // tap 0..26  (kz*9 + ky*3 + kx)
        int co   = nt * 16 + (lane & 15);
        int ci   = (e < 8) ? (e + 8 * (lane >> 4)) : (e + 8 + 8 * (lane >> 4));
        float val = 0.0f;
        if (co < Cout && ci < Cin)
            val = w[(co * Cin + ci) * 27 + t];
        wpk[i] = (__bf16)val;
    }
    if (i < 32) {
        spb[i]      = (i < Cout) ? sc[i] : 0.0f;
        spb[32 + i] = (i < Cout) ? bi[i] : 0.0f;
    }
}

// ---------------------------------------------------------------------------
// m0 = (mask==0); feat0[v][c] = bf16(x[c][v] * m0), channel-padded to 32.
// ---------------------------------------------------------------------------
__global__ void init_input_k(const float* __restrict__ x, const int* __restrict__ mask,
                             __bf16* __restrict__ feat, float* __restrict__ m0, int nvox)
{
    int v = blockIdx.x * blockDim.x + threadIdx.x;
    if (v >= nvox) return;
    float m = (mask[v] == 0) ? 1.0f : 0.0f;
    m0[v] = m;
    #pragma unroll
    for (int c = 0; c < 29; ++c)
        feat[v * 32 + c] = (__bf16)(x[c * nvox + v] * m);
    feat[v * 32 + 29] = (__bf16)0.0f;
    feat[v * 32 + 30] = (__bf16)0.0f;
    feat[v * 32 + 31] = (__bf16)0.0f;
}

// ---------------------------------------------------------------------------
// stride-2 k=3 p=1 mask dilation: active if any active input in the stencil.
// ---------------------------------------------------------------------------
__global__ void dilate_mask_k(const float* __restrict__ mi, float* __restrict__ mo,
                              int resIn, int resOut)
{
    int v = blockIdx.x * blockDim.x + threadIdx.x;
    int n = resOut * resOut * resOut;
    if (v >= n) return;
    int oz = v / (resOut * resOut);
    int oy = (v / resOut) % resOut;
    int ox = v % resOut;
    float any = 0.0f;
    for (int dz = 0; dz < 3; ++dz)
        for (int dy = 0; dy < 3; ++dy)
            for (int dx = 0; dx < 3; ++dx) {
                int iz = 2 * oz + dz - 1, iy = 2 * oy + dy - 1, ix = 2 * ox + dx - 1;
                if ((unsigned)iz < (unsigned)resIn && (unsigned)iy < (unsigned)resIn &&
                    (unsigned)ix < (unsigned)resIn)
                    if (mi[(iz * resIn + iy) * resIn + ix] > 0.0f) any = 1.0f;
            }
    mo[v] = any;
}

// ---------------------------------------------------------------------------
// 3x3x3 conv via WMMA bf16 -> f32. Features: [z][y][x][32] bf16.
// Block = 128 threads = 4 waves; each wave computes 16 output voxels x 32
// output channels. Per tap: one K=32 step = 2x v_wmma_f32_16x16x32_bf16.
// Weights staged in LDS (54 KB). Epilogue: scale/bias + ReLU + mask, bf16 out.
// Output voxel counts (>=4^3=64) are always divisible by 64 -> no tail.
// ---------------------------------------------------------------------------
__global__ __launch_bounds__(128)
void conv3d_bf16_wmma_k(const __bf16* __restrict__ fin,
                        const __bf16* __restrict__ wpk,
                        const float* __restrict__ spb,
                        const float* __restrict__ mout,
                        __bf16* __restrict__ fout,
                        int resIn, int resOut, int stride)
{
    __shared__ __align__(32) __bf16 wlds[27 * 2 * 32 * 16];   // 55296 B

    // cooperative stage of this layer's packed weights into LDS
    {
        const uint4* src = reinterpret_cast<const uint4*>(wpk);
        uint4*       dst = reinterpret_cast<uint4*>(wlds);
        for (int i = threadIdx.x; i < (27 * 2 * 32 * 16 * 2) / 16; i += blockDim.x)
            dst[i] = src[i];
    }
    __syncthreads();

    int lane = threadIdx.x & 31;
    int wave = threadIdx.x >> 5;     // wave32
    int m    = lane & 15;            // A-matrix row (output voxel within tile)
    int hi   = lane >> 4;            // 0: K {0..7,16..23}; 1: K {8..15,24..31}
    int tileBase = (blockIdx.x * 4 + wave) * 16;
    int vox  = tileBase + m;
    int oz = vox / (resOut * resOut);
    int oy = (vox / resOut) % resOut;
    int ox = vox % resOut;

    v8f acc0 = {};
    v8f acc1 = {};
    const v16bf* bw = reinterpret_cast<const v16bf*>(wlds);

    for (int t = 0; t < 27; ++t) {
        int dz = t / 9, dy = (t / 3) % 3, dx = t % 3;
        int iz = oz * stride + dz - 1;
        int iy = oy * stride + dy - 1;
        int ix = ox * stride + dx - 1;

        v16bf a = {};
        if ((unsigned)iz < (unsigned)resIn && (unsigned)iy < (unsigned)resIn &&
            (unsigned)ix < (unsigned)resIn) {
            const __bf16* row = fin + ((iz * resIn + iy) * resIn + ix) * 32;
            v8bf lo = *reinterpret_cast<const v8bf*>(row + (hi << 3));        // ch 0..7 / 8..15
            v8bf hh = *reinterpret_cast<const v8bf*>(row + 16 + (hi << 3));   // ch 16..23 / 24..31
            a = __builtin_shufflevector(lo, hh, 0, 1, 2, 3, 4, 5, 6, 7,
                                                 8, 9, 10, 11, 12, 13, 14, 15);
        }
        v16bf b0 = bw[t * 64 + lane];        // out channels  0..15
        v16bf b1 = bw[t * 64 + 32 + lane];   // out channels 16..31

        acc0 = __builtin_amdgcn_wmma_f32_16x16x32_bf16(false, a, false, b0,
                                                       (short)0, acc0, false, false);
        acc1 = __builtin_amdgcn_wmma_f32_16x16x32_bf16(false, a, false, b1,
                                                       (short)0, acc1, false, false);
    }

    // Epilogue: D layout (f32 16x16): lane = N, VGPR v = row M (lanes>=16: M=v+8)
    int n  = lane & 15;
    float s0  = spb[n],       b0s = spb[32 + n];
    float s1  = spb[16 + n],  b1s = spb[48 + n];
    #pragma unroll
    for (int v2 = 0; v2 < 8; ++v2) {
        int mrow = hi ? (v2 + 8) : v2;
        int ovox = tileBase + mrow;
        float mv = mout[ovox];
        float r0 = fmaxf(acc0[v2] * s0 + b0s, 0.0f) * mv;
        float r1 = fmaxf(acc1[v2] * s1 + b1s, 0.0f) * mv;
        fout[ovox * 32 + n]      = (__bf16)r0;
        fout[ovox * 32 + n + 16] = (__bf16)r1;
    }
}

// ---------------------------------------------------------------------------
// Trilinear grid sample, align_corners=True, zero padding. One thread/point.
// Writes fp32 into out[(chOff+c)*NPTS + p].
// ---------------------------------------------------------------------------
__global__ void grid_sample_k(const __bf16* __restrict__ vol,
                              const float* __restrict__ grid,
                              float* __restrict__ out,
                              int res, int C, int chOff)
{
    int p = blockIdx.x * blockDim.x + threadIdx.x;
    if (p >= GS_NPTS) return;
    float gx = grid[p * 3 + 0], gy = grid[p * 3 + 1], gz = grid[p * 3 + 2];
    float fr = (float)(res - 1);
    float ixf = (gx + 1.0f) * 0.5f * fr;
    float iyf = (gy + 1.0f) * 0.5f * fr;
    float izf = (gz + 1.0f) * 0.5f * fr;
    float x0 = floorf(ixf), y0 = floorf(iyf), z0 = floorf(izf);
    float fx = ixf - x0, fy = iyf - y0, fz = izf - z0;

    float acc[32];
    #pragma unroll
    for (int c = 0; c < 32; ++c) acc[c] = 0.0f;

    for (int dz = 0; dz < 2; ++dz)
        for (int dy = 0; dy < 2; ++dy)
            for (int dx = 0; dx < 2; ++dx) {
                float xc = x0 + dx, yc = y0 + dy, zc = z0 + dz;
                float wx = dx ? fx : (1.0f - fx);
                float wy = dy ? fy : (1.0f - fy);
                float wz = dz ? fz : (1.0f - fz);
                bool valid = (xc >= 0.0f) && (xc <= fr) && (yc >= 0.0f) && (yc <= fr) &&
                             (zc >= 0.0f) && (zc <= fr);
                float w = wx * wy * wz * (valid ? 1.0f : 0.0f);
                int xi = (int)fminf(fmaxf(xc, 0.0f), fr);
                int yi = (int)fminf(fmaxf(yc, 0.0f), fr);
                int zi = (int)fminf(fmaxf(zc, 0.0f), fr);
                const __bf16* row = vol + ((zi * res + yi) * res + xi) * 32;
                #pragma unroll
                for (int c = 0; c < 32; ++c) acc[c] += (float)row[c] * w;
            }

    for (int c = 0; c < C; ++c)
        out[(chOff + c) * GS_NPTS + p] = acc[c];
}

// ---------------------------------------------------------------------------
extern "C" void kernel_launch(void* const* d_in, const int* in_sizes, int n_in,
                              void* d_out, int out_size, void* d_ws, size_t ws_size,
                              hipStream_t stream)
{
    (void)in_sizes; (void)n_in; (void)out_size; (void)ws_size;
    const float* x    = (const float*)d_in[0];
    const float* grid = (const float*)d_in[1];
    const float* W[17]; const float* S[17]; const float* Bp[17];
    for (int i = 0; i < 17; ++i) {
        W[i]  = (const float*)d_in[2 + i];
        S[i]  = (const float*)d_in[19 + i];
        Bp[i] = (const float*)d_in[36 + i];
    }
    const int* mask = (const int*)d_in[53];
    float* out = (float*)d_out;

    char* ws = (char*)d_ws;
    size_t off = 0;
    auto alloc = [&](size_t bytes) -> char* {
        off = (off + 255) & ~(size_t)255;
        char* p = ws + off; off += bytes; return p;
    };
    __bf16* wpk  = (__bf16*)alloc((size_t)17 * 27648 * 2);
    float*  spb  = (float*) alloc((size_t)17 * 64 * 4);
    float*  m0   = (float*) alloc((size_t)262144 * 4);
    float*  m1   = (float*) alloc((size_t)32768 * 4);
    float*  m2   = (float*) alloc((size_t)4096 * 4);
    float*  m3   = (float*) alloc((size_t)512 * 4);
    float*  m4   = (float*) alloc((size_t)64 * 4);
    __bf16* fA64 = (__bf16*)alloc((size_t)262144 * 32 * 2);
    __bf16* fB64 = (__bf16*)alloc((size_t)262144 * 32 * 2);
    __bf16* fA32 = (__bf16*)alloc((size_t)32768 * 32 * 2);
    __bf16* fB32 = (__bf16*)alloc((size_t)32768 * 32 * 2);
    __bf16* fA16 = (__bf16*)alloc((size_t)4096 * 32 * 2);
    __bf16* fB16 = (__bf16*)alloc((size_t)4096 * 32 * 2);
    __bf16* fA8  = (__bf16*)alloc((size_t)512 * 32 * 2);
    __bf16* fB8  = (__bf16*)alloc((size_t)512 * 32 * 2);
    __bf16* fA4  = (__bf16*)alloc((size_t)64 * 32 * 2);
    __bf16* fB4  = (__bf16*)alloc((size_t)64 * 32 * 2);

    static const int Cin[17]  = {29,29,29,29,29,29,29,29,29,29,32,32,32,32,32,32,32};
    static const int Cout[17] = {29,29,29,29,29,29,29,29,29,32,32,32,32,32,32,32,32};
    for (int l = 0; l < 17; ++l)
        pack_weights_k<<<108, 256, 0, stream>>>(W[l], S[l], Bp[l],
                                                wpk + (size_t)l * 27648, spb + l * 64,
                                                Cin[l], Cout[l]);

    init_input_k<<<1024, 256, 0, stream>>>(x, mask, fA64, m0, 262144);
    dilate_mask_k<<<128, 256, 0, stream>>>(m0, m1, 64, 32);
    dilate_mask_k<<<16,  256, 0, stream>>>(m1, m2, 32, 16);
    dilate_mask_k<<<2,   256, 0, stream>>>(m2, m3, 16, 8);
    dilate_mask_k<<<1,   256, 0, stream>>>(m3, m4, 8, 4);

    auto conv = [&](const __bf16* fin, int resIn, __bf16* fout, int resOut,
                    int stride, int layer, const float* mo) {
        int nvox = resOut * resOut * resOut;
        conv3d_bf16_wmma_k<<<nvox / 64, 128, 0, stream>>>(
            fin, wpk + (size_t)layer * 27648, spb + layer * 64, mo, fout,
            resIn, resOut, stride);
    };

    conv(fA64, 64, fB64, 64, 1, 0,  m0);   // conv0 a
    conv(fB64, 64, fA64, 64, 1, 1,  m0);   // conv0 b
    conv(fA64, 64, fA32, 32, 2, 2,  m1);   // down0
    conv(fA32, 32, fB32, 32, 1, 3,  m1);   // conv1 a
    conv(fB32, 32, fA32, 32, 1, 4,  m1);   // conv1 b
    grid_sample_k<<<GS_NPTS / 256, 256, 0, stream>>>(fA32, grid, out, 32, 29, 0);   // f1
    conv(fA32, 32, fA16, 16, 2, 5,  m2);   // down1
    conv(fA16, 16, fB16, 16, 1, 6,  m2);   // conv2 x3
    conv(fB16, 16, fA16, 16, 1, 7,  m2);
    conv(fA16, 16, fB16, 16, 1, 8,  m2);
    grid_sample_k<<<GS_NPTS / 256, 256, 0, stream>>>(fB16, grid, out, 16, 29, 29);  // f2
    conv(fB16, 16, fA8,  8,  2, 9,  m3);   // down2 (29->32)
    conv(fA8,  8,  fB8,  8,  1, 10, m3);   // conv3 x3
    conv(fB8,  8,  fA8,  8,  1, 11, m3);
    conv(fA8,  8,  fB8,  8,  1, 12, m3);
    grid_sample_k<<<GS_NPTS / 256, 256, 0, stream>>>(fB8, grid, out, 8, 32, 58);    // f3
    conv(fB8,  8,  fA4,  4,  2, 13, m4);   // down3
    conv(fA4,  4,  fB4,  4,  1, 14, m4);   // conv4 x3
    conv(fB4,  4,  fA4,  4,  1, 15, m4);
    conv(fA4,  4,  fB4,  4,  1, 16, m4);
    grid_sample_k<<<GS_NPTS / 256, 256, 0, stream>>>(fB4, grid, out, 4, 32, 90);    // f4
}
